// LinearCrossAttention_83133386981927
// MI455X (gfx1250) — matrix operand order repaired and checked by reading, not compiled, not run on hardware
//
#include <hip/hip_runtime.h>
#include <hip/hip_bf16.h>

// ---------------- problem constants ----------------
#define BB 4
#define TT 16384
#define CC 512
#define HH 8
#define DD 64
#define MM (BB * TT)   // 65536 rows

typedef _Float16 half16_t __attribute__((ext_vector_type(16)));
typedef float    float8_t __attribute__((ext_vector_type(8)));
typedef int      int4v    __attribute__((ext_vector_type(4)));

// ------------- gfx1250 async global->LDS copy (guarded) -------------
#if defined(__has_builtin)
#if __has_builtin(__builtin_amdgcn_global_load_async_to_lds_b128)
#define HAVE_ASYNC_LDS 1
#endif
#endif

__device__ __forceinline__ void async_ld128(const void* gptr, void* lptr) {
#ifdef HAVE_ASYNC_LDS
    __builtin_amdgcn_global_load_async_to_lds_b128(
        (__attribute__((address_space(1))) int4v*)gptr,
        (__attribute__((address_space(3))) int4v*)lptr, 0, 0);
#else
    *reinterpret_cast<int4v*>(lptr) = *reinterpret_cast<const int4v*>(gptr);
#endif
}

__device__ __forceinline__ void async_wait() {
#ifdef HAVE_ASYNC_LDS
#if __has_builtin(__builtin_amdgcn_s_wait_asynccnt)
    __builtin_amdgcn_s_wait_asynccnt(0);
#else
    asm volatile("s_wait_asynccnt 0" ::: "memory");
#endif
#endif
}

// A fragment (16x32 f16, M x K): lane m (lo lanes) holds K 0..15,
// hi lanes hold K 16..31. Caller pre-offsets rowbase by +16 for hi lanes.
__device__ __forceinline__ half16_t frag_a(const _Float16* rowbase) {
    half16_t v;
#pragma unroll
    for (int i = 0; i < 16; ++i) v[i] = rowbase[i];
    return v;
}
// B fragment (32x16 f16, K x N): lane n holds 16 contiguous K values
// (lo lanes K 0..15, hi lanes K 16..31). Caller pre-offsets colbase.
__device__ __forceinline__ half16_t frag_b(const _Float16* colbase) {
    half16_t v;
#pragma unroll
    for (int i = 0; i < 16; ++i) v[i] = colbase[i];
    return v;
}

__device__ __forceinline__ float8_t wmma_f16(half16_t a, half16_t b, float8_t c) {
    return __builtin_amdgcn_wmma_f32_16x16x32_f16(false, a, false, b, (short)0, c,
                                                  false, false);
}

// ======================================================================
// Kernel 1: P = softmax_head(X @ W + bias). 128x64 tile per block (one head).
// kmode==0: store P as f16 [bh][t][d] (for q)
// kmode==1: store P transposed [bh][d][T] and atomicAdd column sums into ksum
// ======================================================================
__global__ __launch_bounds__(256) void proj_qk_kernel(
    const float* __restrict__ X, const float* __restrict__ W,
    const float* __restrict__ bias, _Float16* __restrict__ out,
    float* __restrict__ ksum, int kmode)
{
    __shared__ _Float16 As[128][34];
    __shared__ _Float16 Ws[64][34];
    __shared__ float    Cs[128][68];
    __shared__ float    rowscale[128];

    const int tid  = threadIdx.x;
    const int lane = tid & 31;
    const int wave = tid >> 5;
    const bool hi  = lane >= 16;
    const int l16  = lane & 15;

    const int row0  = blockIdx.x * 128;     // global M row
    const int h     = blockIdx.y;
    const int n0    = h * DD;
    const int b     = row0 / TT;
    const int tloc0 = row0 % TT;
    const int bh    = b * HH + h;

    const int wm = (wave & 3) * 32;         // 4 m-blocks of 32 rows
    const int wn = (wave >> 2) * 32;        // 2 n-blocks of 32 cols

    float8_t acc[2][2];
#pragma unroll
    for (int mi = 0; mi < 2; ++mi)
#pragma unroll
        for (int ni = 0; ni < 2; ++ni)
#pragma unroll
            for (int j = 0; j < 8; ++j) acc[mi][ni][j] = 0.f;

    for (int kk = 0; kk < CC; kk += 32) {
        for (int i = tid; i < 128 * 32; i += 256) {
            int r = i >> 5, k = i & 31;
            As[r][k] = (_Float16)X[(size_t)(row0 + r) * CC + kk + k];
            if (kk + 32 < CC)
                __builtin_prefetch(&X[(size_t)(row0 + r) * CC + kk + 32 + k], 0, 0);
        }
        for (int i = tid; i < 32 * 64; i += 256) {
            int k = i >> 6, n = i & 63;
            Ws[n][k] = (_Float16)W[(size_t)(kk + k) * CC + n0 + n];
        }
        __syncthreads();
#pragma unroll
        for (int mi = 0; mi < 2; ++mi) {
            half16_t af = frag_a(&As[wm + mi * 16 + l16][hi ? 16 : 0]);
#pragma unroll
            for (int ni = 0; ni < 2; ++ni) {
                half16_t bf = frag_b(&Ws[wn + ni * 16 + l16][hi ? 16 : 0]);
                acc[mi][ni] = wmma_f16(af, bf, acc[mi][ni]);
            }
        }
        __syncthreads();
    }

    // dump accumulators to LDS (C layout: VGPR r -> row r (+8 for hi lanes))
#pragma unroll
    for (int mi = 0; mi < 2; ++mi)
#pragma unroll
        for (int ni = 0; ni < 2; ++ni)
#pragma unroll
            for (int r = 0; r < 8; ++r)
                Cs[wm + mi * 16 + r + (hi ? 8 : 0)][wn + ni * 16 + l16] = acc[mi][ni][r];
    __syncthreads();

    // softmax over the 64-wide head dimension, one thread per row
    if (tid < 128) {
        float mx = -1e30f;
        for (int d = 0; d < DD; ++d) {
            float v = Cs[tid][d] + bias[n0 + d];
            Cs[tid][d] = v;
            mx = fmaxf(mx, v);
        }
        float s = 0.f;
        for (int d = 0; d < DD; ++d) {
            float e = __expf(Cs[tid][d] - mx);
            Cs[tid][d] = e;
            s += e;
        }
        rowscale[tid] = 1.f / s;
    }
    __syncthreads();

    if (kmode && tid < DD) {           // k_cumsum accumulation (per bh, per d)
        float s = 0.f;
        for (int r = 0; r < 128; ++r) s += Cs[r][tid] * rowscale[r];
        atomicAdd(&ksum[bh * DD + tid], s);
    }

    if (!kmode) {                      // q: [bh][t][d]
        for (int i = tid; i < 128 * DD; i += 256) {
            int r = i >> 6, d = i & 63;
            out[((size_t)bh * TT + tloc0 + r) * DD + d] =
                (_Float16)(Cs[r][d] * rowscale[r]);
        }
    } else {                           // k: transposed [bh][d][T]
        for (int i = tid; i < DD * 128; i += 256) {
            int d = i >> 7, t = i & 127;
            out[((size_t)bh * DD + d) * TT + tloc0 + t] =
                (_Float16)(Cs[t][d] * rowscale[t]);
        }
    }
}

// ======================================================================
// Kernel 2: v = y0@Wv0 + y1@Wv1 + (bv0+bv1), stored transposed [bh][d][T]
// ======================================================================
__global__ __launch_bounds__(256) void proj_v_kernel(
    const float* __restrict__ y0, const float* __restrict__ y1,
    const float* __restrict__ Wv, const float* __restrict__ bv,
    _Float16* __restrict__ vT)
{
    __shared__ _Float16 As[128][34];
    __shared__ _Float16 Ws[64][34];
    __shared__ float    Cs[128][68];

    const int tid  = threadIdx.x;
    const int lane = tid & 31;
    const int wave = tid >> 5;
    const bool hi  = lane >= 16;
    const int l16  = lane & 15;

    const int row0  = blockIdx.x * 128;
    const int h     = blockIdx.y;
    const int n0    = h * DD;
    const int b     = row0 / TT;
    const int tloc0 = row0 % TT;
    const int bh    = b * HH + h;

    const int wm = (wave & 3) * 32;
    const int wn = (wave >> 2) * 32;

    float8_t acc[2][2];
#pragma unroll
    for (int mi = 0; mi < 2; ++mi)
#pragma unroll
        for (int ni = 0; ni < 2; ++ni)
#pragma unroll
            for (int j = 0; j < 8; ++j) acc[mi][ni][j] = 0.f;

    for (int kk = 0; kk < 2 * CC; kk += 32) {  // virtual K = 1024: [y0 y1] @ [Wv0;Wv1]
        const float* Y = (kk < CC) ? y0 : y1;
        const size_t wbase = (kk < CC) ? 0 : (size_t)CC * CC;
        const int kc = kk & (CC - 1);
        for (int i = tid; i < 128 * 32; i += 256) {
            int r = i >> 5, k = i & 31;
            As[r][k] = (_Float16)Y[(size_t)(row0 + r) * CC + kc + k];
            if (kc + 32 < CC)
                __builtin_prefetch(&Y[(size_t)(row0 + r) * CC + kc + 32 + k], 0, 0);
        }
        for (int i = tid; i < 32 * 64; i += 256) {
            int k = i >> 6, n = i & 63;
            Ws[n][k] = (_Float16)Wv[wbase + (size_t)(kc + k) * CC + n0 + n];
        }
        __syncthreads();
#pragma unroll
        for (int mi = 0; mi < 2; ++mi) {
            half16_t af = frag_a(&As[wm + mi * 16 + l16][hi ? 16 : 0]);
#pragma unroll
            for (int ni = 0; ni < 2; ++ni) {
                half16_t bf = frag_b(&Ws[wn + ni * 16 + l16][hi ? 16 : 0]);
                acc[mi][ni] = wmma_f16(af, bf, acc[mi][ni]);
            }
        }
        __syncthreads();
    }

#pragma unroll
    for (int mi = 0; mi < 2; ++mi)
#pragma unroll
        for (int ni = 0; ni < 2; ++ni)
#pragma unroll
            for (int r = 0; r < 8; ++r)
                Cs[wm + mi * 16 + r + (hi ? 8 : 0)][wn + ni * 16 + l16] = acc[mi][ni][r];
    __syncthreads();

    for (int i = tid; i < DD * 128; i += 256) {   // transposed store [bh][d][T]
        int d = i >> 7, t = i & 127;
        float val = Cs[t][d] + bv[n0 + d] + bv[CC + n0 + d];
        vT[((size_t)bh * DD + d) * TT + tloc0 + t] = (_Float16)val;
    }
}

// ======================================================================
// Kernel 3: kv[d][e] = sum_t k[t][d]*v[t][e]
// grid = (T/2048 windows, 32 bh); split-K atomicAdd into f32 kv.
// kT/vT tiles staged via async global->LDS (B128) copies.
// ======================================================================
__global__ __launch_bounds__(256) void kv_kernel(
    const _Float16* __restrict__ kT, const _Float16* __restrict__ vT,
    float* __restrict__ kv)
{
    __shared__ _Float16 Ks[64][136];   // 272B rows: 16B-aligned, conflict-free
    __shared__ _Float16 Vs[64][136];

    const int tid  = threadIdx.x;
    const int lane = tid & 31;
    const int wave = tid >> 5;
    const bool hi  = lane >= 16;
    const int l16  = lane & 15;

    const int bh   = blockIdx.y;
    const int t0   = blockIdx.x * 2048;
    const int dblk = (wave & 3) * 16;      // M block (d)
    const int e0   = (wave >> 2) * 32;     // N block base (e), 2 tiles each

    float8_t acc[2];
#pragma unroll
    for (int ni = 0; ni < 2; ++ni)
#pragma unroll
        for (int j = 0; j < 8; ++j) acc[ni][j] = 0.f;

    for (int tc = 0; tc < 2048; tc += 128) {
        // 64 rows x 128 halves = 64 rows x 16 chunks of 16B, per array
        for (int c = tid; c < 64 * 16; c += 256) {
            int d = c >> 4, ch = c & 15;
            size_t g = ((size_t)bh * DD + d) * TT + t0 + tc + ch * 8;
            async_ld128(&kT[g], &Ks[d][ch * 8]);
            async_ld128(&vT[g], &Vs[d][ch * 8]);
        }
        async_wait();
        __syncthreads();
#pragma unroll
        for (int ks = 0; ks < 128; ks += 32) {
            half16_t af = frag_a(&Ks[dblk + l16][ks + (hi ? 16 : 0)]);
#pragma unroll
            for (int ni = 0; ni < 2; ++ni) {
                half16_t bf = frag_b(&Vs[e0 + ni * 16 + l16][ks + (hi ? 16 : 0)]);
                acc[ni] = wmma_f16(af, bf, acc[ni]);
            }
        }
        __syncthreads();
    }
#pragma unroll
    for (int ni = 0; ni < 2; ++ni)
#pragma unroll
        for (int r = 0; r < 8; ++r)
            atomicAdd(&kv[((size_t)bh * DD + dblk + r + (hi ? 8 : 0)) * DD +
                          e0 + ni * 16 + l16],
                      acc[ni][r]);
}

// ======================================================================
// Kernel 4: out_h = (q @ kv) * (1 / (q . ksum)), stored f16 [B,T,C]
// grid = (T/128, 32 bh). q tile staged via async copies.
// ======================================================================
__global__ __launch_bounds__(256) void attn_kernel(
    const _Float16* __restrict__ q, const float* __restrict__ kv,
    const float* __restrict__ ksum, _Float16* __restrict__ attn)
{
    __shared__ _Float16 Qs[128][72];   // 144B rows: 16B-aligned, conflict-free
    __shared__ _Float16 KVs[64][68];
    __shared__ float ksums[64];
    __shared__ float dinv[128];

    const int tid  = threadIdx.x;
    const int lane = tid & 31;
    const int wave = tid >> 5;
    const bool hi  = lane >= 16;
    const int l16  = lane & 15;

    const int bh = blockIdx.y;
    const int b  = bh / HH;
    const int h  = bh % HH;
    const int t0 = blockIdx.x * 128;

    // 128 rows x 64 halves = 128 rows x 8 chunks of 16B
    for (int c = tid; c < 128 * 8; c += 256) {
        int r = c >> 3, ch = c & 7;
        async_ld128(&q[((size_t)bh * TT + t0 + r) * DD + ch * 8], &Qs[r][ch * 8]);
    }
    for (int i = tid; i < DD * DD; i += 256) {
        int d = i >> 6, e = i & 63;
        KVs[d][e] = (_Float16)kv[((size_t)bh * DD + d) * DD + e];
    }
    if (tid < DD) ksums[tid] = ksum[bh * DD + tid];
    async_wait();
    __syncthreads();

    if (tid < 128) {
        float s = 0.f;
        for (int d = 0; d < DD; ++d) s += (float)Qs[tid][d] * ksums[d];
        dinv[tid] = 1.f / s;
    }
    __syncthreads();

    const int wt = wave * 16;            // each wave owns one 16-row t block
    float8_t acc[4];
#pragma unroll
    for (int eb = 0; eb < 4; ++eb)
#pragma unroll
        for (int j = 0; j < 8; ++j) acc[eb][j] = 0.f;

#pragma unroll
    for (int dk = 0; dk < DD; dk += 32) {
        half16_t af = frag_a(&Qs[wt + l16][dk + (hi ? 16 : 0)]);
#pragma unroll
        for (int eb = 0; eb < 4; ++eb) {
            half16_t bf;
            const int e  = eb * 16 + l16;
            const int kb = dk + (hi ? 16 : 0);
#pragma unroll
            for (int i = 0; i < 16; ++i) bf[i] = KVs[kb + i][e];
            acc[eb] = wmma_f16(af, bf, acc[eb]);
        }
    }

#pragma unroll
    for (int eb = 0; eb < 4; ++eb)
#pragma unroll
        for (int r = 0; r < 8; ++r) {
            int row = wt + r + (hi ? 8 : 0);
            float v = acc[eb][r] * dinv[row];
            attn[((size_t)b * TT + t0 + row) * CC + h * DD + eb * 16 + l16] =
                (_Float16)v;
        }
}

// ======================================================================
// Kernel 5: out = attn @ Wp + bp   (f16 A staged async, f32 output)
// ======================================================================
__global__ __launch_bounds__(256) void out_proj_kernel(
    const _Float16* __restrict__ A, const float* __restrict__ Wp,
    const float* __restrict__ bp, float* __restrict__ out)
{
    __shared__ _Float16 As[128][40];   // 80B rows: 16B-aligned, conflict-free
    __shared__ _Float16 Ws[64][34];

    const int tid  = threadIdx.x;
    const int lane = tid & 31;
    const int wave = tid >> 5;
    const bool hi  = lane >= 16;
    const int l16  = lane & 15;

    const int row0 = blockIdx.x * 128;
    const int n0   = blockIdx.y * DD;

    const int wm = (wave & 3) * 32;
    const int wn = (wave >> 2) * 32;

    float8_t acc[2][2];
#pragma unroll
    for (int mi = 0; mi < 2; ++mi)
#pragma unroll
        for (int ni = 0; ni < 2; ++ni)
#pragma unroll
            for (int j = 0; j < 8; ++j) acc[mi][ni][j] = 0.f;

    for (int kk = 0; kk < CC; kk += 32) {
        // 128 rows x 32 halves = 128 rows x 4 chunks of 16B
        for (int c = tid; c < 128 * 4; c += 256) {
            int r = c >> 2, ch = c & 3;
            async_ld128(&A[(size_t)(row0 + r) * CC + kk + ch * 8], &As[r][ch * 8]);
        }
        for (int i = tid; i < 32 * 64; i += 256) {
            int k = i >> 6, n = i & 63;
            Ws[n][k] = (_Float16)Wp[(size_t)(kk + k) * CC + n0 + n];
        }
        async_wait();
        __syncthreads();
#pragma unroll
        for (int mi = 0; mi < 2; ++mi) {
            half16_t af = frag_a(&As[wm + mi * 16 + l16][hi ? 16 : 0]);
#pragma unroll
            for (int ni = 0; ni < 2; ++ni) {
                half16_t bf = frag_b(&Ws[wn + ni * 16 + l16][hi ? 16 : 0]);
                acc[mi][ni] = wmma_f16(af, bf, acc[mi][ni]);
            }
        }
        __syncthreads();
    }

#pragma unroll
    for (int mi = 0; mi < 2; ++mi)
#pragma unroll
        for (int ni = 0; ni < 2; ++ni)
#pragma unroll
            for (int r = 0; r < 8; ++r) {
                int row = row0 + wm + mi * 16 + r + (hi ? 8 : 0);
                int col = n0 + wn + ni * 16 + l16;
                out[(size_t)row * CC + col] = acc[mi][ni][r] + bp[col];
            }
}

// ======================================================================
extern "C" void kernel_launch(void* const* d_in, const int* in_sizes, int n_in,
                              void* d_out, int out_size, void* d_ws, size_t ws_size,
                              hipStream_t stream)
{
    (void)in_sizes; (void)n_in; (void)out_size; (void)ws_size;

    const float* x1 = (const float*)d_in[0];
    const float* x2 = (const float*)d_in[1];
    const float* y0 = (const float*)d_in[2];
    const float* y1 = (const float*)d_in[3];
    const float* Wq = (const float*)d_in[4];
    const float* bq = (const float*)d_in[5];
    const float* Wk = (const float*)d_in[6];
    const float* bk = (const float*)d_in[7];
    const float* Wv = (const float*)d_in[8];
    const float* bv = (const float*)d_in[9];
    const float* Wp = (const float*)d_in[10];
    const float* bp = (const float*)d_in[11];
    float* out = (float*)d_out;

    char* ws = (char*)d_ws;
    size_t off = 0;
    const size_t f16mat = (size_t)MM * CC * sizeof(_Float16);   // 64 MiB each
    _Float16* qh   = (_Float16*)(ws + off); off += f16mat;
    _Float16* kT   = (_Float16*)(ws + off); off += f16mat;
    _Float16* vT   = (_Float16*)(ws + off); off += f16mat;
    _Float16* attn = (_Float16*)(ws + off); off += f16mat;
    float* ksum = (float*)(ws + off); off += (size_t)BB * HH * DD * sizeof(float);
    float* kvb  = (float*)(ws + off); off += (size_t)BB * HH * DD * DD * sizeof(float);

    (void)hipMemsetAsync(ksum, 0, (size_t)BB * HH * DD * sizeof(float), stream);
    (void)hipMemsetAsync(kvb, 0, (size_t)BB * HH * DD * DD * sizeof(float), stream);

    dim3 blk(256);
    proj_qk_kernel<<<dim3(MM / 128, HH), blk, 0, stream>>>(x1, Wq, bq, qh, nullptr, 0);
    proj_qk_kernel<<<dim3(MM / 128, HH), blk, 0, stream>>>(x2, Wk, bk, kT, ksum, 1);
    proj_v_kernel<<<dim3(MM / 128, HH), blk, 0, stream>>>(y0, y1, Wv, bv, vT);
    kv_kernel<<<dim3(TT / 2048, BB * HH), blk, 0, stream>>>(kT, vT, kvb);
    attn_kernel<<<dim3(TT / 128, BB * HH), blk, 0, stream>>>(qh, kvb, ksum, attn);
    out_proj_kernel<<<dim3(MM / 128, HH), blk, 0, stream>>>(attn, Wp, bp, out);
}
